// BigBirdSeq2SeqDecoder_69020124446912
// MI455X (gfx1250) — compile-verified, matrix operand order checked by reading, not run on hardware
//
#include <hip/hip_runtime.h>
#include <cstdint>
#include <cstddef>

// ---------------- problem constants ----------------
#define B_    2
#define S_    1024
#define E_    512
#define H_    8
#define Dh_   64
#define HD_   512          // H*Dh
#define M_    2048
#define V_    32000
#define L_    2
#define NB_   16           // S/BLOCK
#define ROWS_ (B_*S_)      // 2048
#define NEGV  (-1.0e9f)

typedef __bf16        v16bf __attribute__((ext_vector_type(16)));
typedef float         v8f   __attribute__((ext_vector_type(8)));
typedef unsigned int  v8u   __attribute__((ext_vector_type(8)));
typedef unsigned short u16;

// float -> bf16 bits (round to nearest even)
__device__ __forceinline__ u16 f2bf(float f) {
    unsigned int u = __float_as_uint(f);
    u += 0x7FFFu + ((u >> 16) & 1u);
    return (u16)(u >> 16);
}

// K index inside a 32-wide chunk for 16-bit A/B fragments (ISA 7.12.2):
// VGPR g in 0..7, lane-half kh: K = 16*(g/4) + 2*(g%4) + 8*kh  (+pair)
__device__ __forceinline__ int frag_k(int g, int kh) {
    return ((g >> 2) << 4) + ((g & 3) << 1) + (kh << 3);
}

// swizzled flat LDS tile: row stride 32 bf16 (64B), 16B chunks XORed by (row>>2)&3.
// returns pointer to the 4B pair at (row, k) -- k even, pair stays inside a chunk.
__device__ __forceinline__ const unsigned int* frag_pair(const u16* base, int row, int k) {
    int sw = (row >> 2) & 3;
    int off = row * 32 + (((k >> 3) ^ sw) << 3) + (k & 7);
    return (const unsigned int*)(base + off);
}

// ---- CDNA5 async global->LDS (ASYNCcnt path), 16B per lane -----------------
__device__ __forceinline__ void async_b128(unsigned lds_off, const void* gptr) {
    unsigned long long ga = (unsigned long long)(uintptr_t)gptr;
    asm volatile("global_load_async_to_lds_b128 %0, %1, off"
                 :: "v"(lds_off), "v"(ga) : "memory");
}
__device__ __forceinline__ void wait_async0() {
    asm volatile("s_wait_asynccnt 0" ::: "memory");
}

// ---------------- MT19937 (matches numpy legacy RandomState, scalar seed) ---
#define MTN 624
struct MTState { unsigned int mt[MTN]; int idx; };

__device__ void mt_seed(MTState* st, unsigned int s) {
    st->mt[0] = s;
    for (int i = 1; i < MTN; ++i)
        st->mt[i] = 1812433253u * (st->mt[i-1] ^ (st->mt[i-1] >> 30)) + (unsigned)i;
    st->idx = MTN;
}
__device__ unsigned int mt_next(MTState* st) {
    if (st->idx >= MTN) {
        for (int i = 0; i < MTN; ++i) {
            unsigned int y = (st->mt[i] & 0x80000000u) | (st->mt[(i+1)%MTN] & 0x7fffffffu);
            unsigned int v = st->mt[(i+397)%MTN] ^ (y >> 1);
            if (y & 1u) v ^= 2567483615u;
            st->mt[i] = v;
        }
        st->idx = 0;
    }
    unsigned int y = st->mt[st->idx++];
    y ^= y >> 11; y ^= (y << 7) & 2636928640u; y ^= (y << 15) & 4022730752u; y ^= y >> 18;
    return y;
}
__device__ unsigned int rk_interval(MTState* st, unsigned int mx) {
    if (mx == 0u) return 0u;
    unsigned int mask = mx;
    mask |= mask >> 1; mask |= mask >> 2; mask |= mask >> 4; mask |= mask >> 8; mask |= mask >> 16;
    unsigned int v;
    while ((v = (mt_next(st) & mask)) > mx) {}
    return v;
}

// builds 4 block masks: self L0 (seed 1000), self L1 (1001), cross L0 (0), cross L1 (1)
__global__ void build_masks_kernel(unsigned char* __restrict__ masks) {
    if (threadIdx.x != 0 || blockIdx.x != 0) return;
    const int seeds[4] = {1000, 1001, 0, 1};
    for (int mi = 0; mi < 4; ++mi) {
        unsigned char* m = masks + mi * NB_ * NB_;
        for (int i = 0; i < NB_*NB_; ++i) m[i] = 0;
        for (int i = 0; i < NB_; ++i)
            for (int off = -1; off <= 1; ++off) {
                int j = i + off;
                if (j >= 0 && j < NB_) m[i*NB_ + j] = 1;
            }
        for (int j = 0; j < NB_; ++j) { m[0*NB_ + j] = 1; m[(NB_-1)*NB_ + j] = 1; }
        for (int i = 0; i < NB_; ++i) { m[i*NB_ + 0] = 1; m[i*NB_ + NB_-1] = 1; }
        MTState st; mt_seed(&st, (unsigned)seeds[mi]);
        for (int i = 0; i < NB_; ++i) {
            int perm[NB_];
            for (int t = 0; t < NB_; ++t) perm[t] = t;
            for (int t = NB_-1; t > 0; --t) {
                unsigned int j = rk_interval(&st, (unsigned)t);
                int tmp = perm[t]; perm[t] = perm[j]; perm[j] = tmp;
            }
            for (int c = 0; c < 3; ++c) m[i*NB_ + perm[c]] = 1;
        }
    }
}

// ---------------- conversion kernels ----------------------------------------
__global__ __launch_bounds__(256)
void cvt_kernel(const float* __restrict__ in, u16* __restrict__ out, int n) {
    int i = blockIdx.x * 256 + threadIdx.x;
    if (i < n) out[i] = f2bf(in[i]);
}
// in: [K][N] f32 row-major; out: [N][K] bf16 (pre-transposed GEMM B operand)
__global__ __launch_bounds__(256)
void cvt_t_kernel(const float* __restrict__ in, u16* __restrict__ out, int K, int N) {
    int i = blockIdx.x * 256 + threadIdx.x;
    if (i < N * K) {
        int n = i / K, k = i - n * K;
        out[i] = f2bf(in[(size_t)k * N + n]);
    }
}

// ---------------- embedding + positional ------------------------------------
__global__ __launch_bounds__(128)
void embed_kernel(const int* __restrict__ targets, const float* __restrict__ table,
                  const float* __restrict__ pos, float* __restrict__ Y) {
    int bs = blockIdx.x;
    int b = bs >> 10, s = bs & 1023;
    int t = (s == 0) ? 0 : targets[b*S_ + s - 1];
    const float* tr = table + (size_t)t * E_;
    const float* pr = pos + (size_t)s * E_;
    float* yr = Y + (size_t)bs * E_;
    for (int e = threadIdx.x; e < E_; e += 128) yr[e] = tr[e] + pr[e];
}

// ---------------- layernorm (f32 in -> bf16 out) ----------------------------
__global__ __launch_bounds__(128)
void layernorm_kernel(const float* __restrict__ X, u16* __restrict__ Y,
                      const float* __restrict__ sc, const float* __restrict__ bi) {
    __shared__ float red[128];
    const int row = blockIdx.x, tid = threadIdx.x;
    const float* xr = X + (size_t)row * E_;
    float sum = 0.f, sq = 0.f;
    for (int e = tid; e < E_; e += 128) { float v = xr[e]; sum += v; sq += v*v; }
    red[tid] = sum; __syncthreads();
    for (int st = 64; st > 0; st >>= 1) { if (tid < st) red[tid] += red[tid+st]; __syncthreads(); }
    float mean = red[0] * (1.0f / E_);
    __syncthreads();
    red[tid] = sq; __syncthreads();
    for (int st = 64; st > 0; st >>= 1) { if (tid < st) red[tid] += red[tid+st]; __syncthreads(); }
    float var = red[0] * (1.0f / E_) - mean * mean;
    float rstd = rsqrtf(var + 1e-6f);
    u16* yr = Y + (size_t)row * E_;
    for (int e = tid; e < E_; e += 128) yr[e] = f2bf((xr[e] - mean) * rstd * sc[e] + bi[e]);
}

// ---------------- bf16 WMMA GEMM (double-buffered async staging) ------------
// A: MxK bf16 row-major. Bt: NxK bf16 (pre-transposed). C: MxN f32 or bf16.
// C = [relu]( alpha*A*B [+ bias] [+ res] ).
#define BM 128
#define BN 128
#define BK 32
#define TILE_U16 (BM * BK)     // one A (or B) tile in u16 elements

// issue one k-step's A+B tiles into LDS buffer (async, 16B chunks, swizzled)
__device__ __forceinline__ void stage_tiles(const u16* __restrict__ A,
                                            const u16* __restrict__ Bt,
                                            unsigned asOff, unsigned bsOff,
                                            int tid, int tileM, int tileN,
                                            int K, int k0) {
    #pragma unroll
    for (int i = 0; i < 2; ++i) {
        int idx = tid + i * 256;              // 0..511 chunks of 16B
        int m = idx >> 2, qc = idx & 3;
        int sw = qc ^ ((m >> 2) & 3);
        async_b128(asOff + (unsigned)((m * 32 + (sw << 3)) * 2),
                   A + (size_t)(tileM + m) * K + k0 + qc * 8);
    }
    #pragma unroll
    for (int i = 0; i < 2; ++i) {
        int idx = tid + i * 256;
        int n = idx >> 2, qc = idx & 3;
        int sw = qc ^ ((n >> 2) & 3);
        async_b128(bsOff + (unsigned)((n * 32 + (sw << 3)) * 2),
                   Bt + (size_t)(tileN + n) * K + k0 + qc * 8);
    }
}

template<bool HAS_BIAS, bool HAS_RES, bool RELU, bool OUTBF>
__global__ __launch_bounds__(256)
void gemm_kernel(const u16* __restrict__ A, const u16* __restrict__ Bt,
                 void* __restrict__ Cv, const float* __restrict__ bias,
                 const float* __restrict__ res, int M, int N, int K, float alpha) {
    __shared__ __align__(16) u16 As[2][TILE_U16];
    __shared__ __align__(16) u16 Bs[2][TILE_U16];
    const int tid  = threadIdx.x;
    const int tileM = blockIdx.y * BM, tileN = blockIdx.x * BN;
    const int w = tid >> 5, lane = tid & 31;
    const int wm = w & 3, wn = w >> 2;            // 4x2 wave grid, wave tile 32x64
    const int ln16 = lane & 15, kh = lane >> 4;
    const unsigned asOff0 = (unsigned)(uintptr_t)(void*)&As[0][0];
    const unsigned bsOff0 = (unsigned)(uintptr_t)(void*)&Bs[0][0];

    v8f c[2][4];
    #pragma unroll
    for (int tm = 0; tm < 2; ++tm)
        #pragma unroll
        for (int tn = 0; tn < 4; ++tn)
            #pragma unroll
            for (int r = 0; r < 8; ++r) c[tm][tn][r] = 0.f;

    const int T = K / BK;
    // prologue: stage tile 0 into buffer 0
    stage_tiles(A, Bt, asOff0, bsOff0, tid, tileM, tileN, K, 0);

    for (int t = 0; t < T; ++t) {
        const int cur = t & 1, nxt = cur ^ 1;
        wait_async0();          // this wave's outstanding async writes done
        __syncthreads();        // all waves' tile-t data visible in LDS
        if (t + 1 < T)          // overlap: stage tile t+1 while computing tile t
            stage_tiles(A, Bt,
                        asOff0 + (unsigned)(nxt * TILE_U16 * 2),
                        bsOff0 + (unsigned)(nxt * TILE_U16 * 2),
                        tid, tileM, tileN, K, (t + 1) * BK);

        const u16* __restrict__ Asb = &As[cur][0];
        const u16* __restrict__ Bsb = &Bs[cur][0];
        v16bf af[2], bfv[4];
        #pragma unroll
        for (int tm = 0; tm < 2; ++tm) {
            int row = wm * 32 + tm * 16 + ln16;
            v8u tt;
            #pragma unroll
            for (int g = 0; g < 8; ++g)
                tt[g] = *frag_pair(Asb, row, frag_k(g, kh));
            af[tm] = __builtin_bit_cast(v16bf, tt);
        }
        #pragma unroll
        for (int tn = 0; tn < 4; ++tn) {
            int col = wn * 64 + tn * 16 + ln16;
            v8u tt;
            #pragma unroll
            for (int g = 0; g < 8; ++g)
                tt[g] = *frag_pair(Bsb, col, frag_k(g, kh));
            bfv[tn] = __builtin_bit_cast(v16bf, tt);
        }
        #pragma unroll
        for (int tm = 0; tm < 2; ++tm)
            #pragma unroll
            for (int tn = 0; tn < 4; ++tn)
                c[tm][tn] = __builtin_amdgcn_wmma_f32_16x16x32_bf16(
                    false, af[tm], false, bfv[tn], (short)0, c[tm][tn], false, false);
        __syncthreads();        // protect buffer 'cur' before it is re-staged at t+2
    }

    // epilogue: compile-time specialized, bias hoisted
    float bv[4] = {0.f, 0.f, 0.f, 0.f};
    if constexpr (HAS_BIAS) {
        #pragma unroll
        for (int tn = 0; tn < 4; ++tn)
            bv[tn] = bias[tileN + wn * 64 + tn * 16 + ln16];
    }
    #pragma unroll
    for (int tm = 0; tm < 2; ++tm)
        #pragma unroll
        for (int tn = 0; tn < 4; ++tn)
            #pragma unroll
            for (int r = 0; r < 8; ++r) {
                int row = tileM + wm * 32 + tm * 16 + kh * 8 + r;
                int col = tileN + wn * 64 + tn * 16 + ln16;
                size_t o = (size_t)row * N + col;
                float v = c[tm][tn][r] * alpha + bv[tn];
                if constexpr (HAS_RES) v += res[o];
                if constexpr (RELU)    v = fmaxf(v, 0.f);
                if constexpr (OUTBF)   ((u16*)Cv)[o] = f2bf(v);
                else                   ((float*)Cv)[o] = v;
            }
}

// ---------------- fused block-sparse flash attention ------------------------
// grid (NB_, H_, B_), 128 threads = 4 waves; each wave owns 16 q rows.
// Q/K/V/O are bf16 [B*S, HD].
__global__ __launch_bounds__(128)
void attn_kernel(const u16* __restrict__ Q, const u16* __restrict__ Kp,
                 const u16* __restrict__ Vp, u16* __restrict__ O,
                 const unsigned char* __restrict__ bmask,
                 const int* __restrict__ targets,
                 const float* __restrict__ srcmask,   // null => k-pad from targets
                 int causal) {
    const int qb = blockIdx.x, h = blockIdx.y, b = blockIdx.z;
    __shared__ __align__(16) u16 Qs[64][66];      // [q][d]
    __shared__ __align__(16) u16 Ks[64][66];      // [k][d]
    __shared__ __align__(16) u16 Vs[64][66];      // [d][k] (transposed)
    __shared__ __align__(16) u16 Ps[4][16][66];   // per-wave P tile [q][k]
    __shared__ unsigned char kdead[64];

    const int tid = threadIdx.x, w = tid >> 5, lane = tid & 31;
    const int ln16 = lane & 15, kh = lane >> 4;

    #pragma unroll
    for (int i = 0; i < 16; ++i) {                 // stage Q tile (bf16 pairs)
        int idx = tid + i * 128;                   // 0..2047 dword chunks
        int qi = idx >> 5, dp = (idx & 31) << 1;
        int row = b * S_ + qb * 64 + qi;
        *(unsigned int*)&Qs[qi][dp] =
            *(const unsigned int*)&Q[(size_t)row * HD_ + h * Dh_ + dp];
    }
    __syncthreads();

    v16bf aq[2];
    #pragma unroll
    for (int f = 0; f < 2; ++f) {
        v8u t;
        #pragma unroll
        for (int g = 0; g < 8; ++g)
            t[g] = *(const unsigned int*)&Qs[w * 16 + ln16][f * 32 + frag_k(g, kh)];
        aq[f] = __builtin_bit_cast(v16bf, t);
    }

    v8f oc[4];
    float m_r[8], l_r[8];
    #pragma unroll
    for (int dn = 0; dn < 4; ++dn)
        #pragma unroll
        for (int r = 0; r < 8; ++r) oc[dn][r] = 0.f;
    #pragma unroll
    for (int r = 0; r < 8; ++r) { m_r[r] = -INFINITY; l_r[r] = 0.f; }

    for (int kb = 0; kb < NB_; ++kb) {
        if (causal && kb > qb) continue;
        if (!bmask[qb * NB_ + kb]) continue;
        __syncthreads();
        #pragma unroll
        for (int i = 0; i < 16; ++i) {             // stage K (direct) and V (transposed)
            int idx = tid + i * 128;
            int ki = idx >> 5, dp = (idx & 31) << 1;
            int row = b * S_ + kb * 64 + ki;
            *(unsigned int*)&Ks[ki][dp] =
                *(const unsigned int*)&Kp[(size_t)row * HD_ + h * Dh_ + dp];
            unsigned int wv = *(const unsigned int*)&Vp[(size_t)row * HD_ + h * Dh_ + dp];
            Vs[dp][ki]     = (u16)(wv & 0xffffu);
            Vs[dp + 1][ki] = (u16)(wv >> 16);
        }
        if (tid < 64) {
            int kg = kb * 64 + tid;
            bool ok = srcmask ? (srcmask[b * S_ + kg] > 0.f) : (targets[b * S_ + kg] > 0);
            kdead[tid] = ok ? 0 : 1;
        }
        __syncthreads();

        // S = Q * K^T
        v8f s[4];
        #pragma unroll
        for (int tn = 0; tn < 4; ++tn) {
            v16bf bk0, bk1;
            { v8u t;
              #pragma unroll
              for (int g = 0; g < 8; ++g)
                  t[g] = *(const unsigned int*)&Ks[tn * 16 + ln16][frag_k(g, kh)];
              bk0 = __builtin_bit_cast(v16bf, t); }
            { v8u t;
              #pragma unroll
              for (int g = 0; g < 8; ++g)
                  t[g] = *(const unsigned int*)&Ks[tn * 16 + ln16][32 + frag_k(g, kh)];
              bk1 = __builtin_bit_cast(v16bf, t); }
            v8f acc;
            #pragma unroll
            for (int r = 0; r < 8; ++r) acc[r] = 0.f;
            acc = __builtin_amdgcn_wmma_f32_16x16x32_bf16(false, aq[0], false, bk0, (short)0, acc, false, false);
            acc = __builtin_amdgcn_wmma_f32_16x16x32_bf16(false, aq[1], false, bk1, (short)0, acc, false, false);
            s[tn] = acc;
        }

        // mask + online softmax
        #pragma unroll
        for (int r = 0; r < 8; ++r) {
            int qlocal = w * 16 + kh * 8 + r;
            int qg = qb * 64 + qlocal;
            bool qp = targets[b * S_ + qg] > 0;
            float sv[4];
            #pragma unroll
            for (int tn = 0; tn < 4; ++tn) {
                int klocal = tn * 16 + ln16;
                int kg = kb * 64 + klocal;
                bool allow = qp && (kdead[klocal] == 0) && (!causal || kg <= qg);
                sv[tn] = allow ? s[tn][r] : NEGV;
            }
            float mx = fmaxf(fmaxf(sv[0], sv[1]), fmaxf(sv[2], sv[3]));
            #pragma unroll
            for (int sx = 1; sx < 16; sx <<= 1) mx = fmaxf(mx, __shfl_xor(mx, sx, 32));
            float mnew = fmaxf(m_r[r], mx);
            float scale = __expf(m_r[r] - mnew);
            float rs = 0.f;
            #pragma unroll
            for (int tn = 0; tn < 4; ++tn) {
                float p = __expf(sv[tn] - mnew);
                rs += p;
                Ps[w][kh * 8 + r][tn * 16 + ln16] = f2bf(p);
            }
            #pragma unroll
            for (int sx = 1; sx < 16; sx <<= 1) rs += __shfl_xor(rs, sx, 32);
            l_r[r] = l_r[r] * scale + rs;
            m_r[r] = mnew;
            #pragma unroll
            for (int tn = 0; tn < 4; ++tn) oc[tn][r] *= scale;
        }

        // O += P * V  (per-wave LDS; DS ops in-order within a wave)
        v16bf ap0, ap1;
        { v8u t;
          #pragma unroll
          for (int g = 0; g < 8; ++g)
              t[g] = *(const unsigned int*)&Ps[w][ln16][frag_k(g, kh)];
          ap0 = __builtin_bit_cast(v16bf, t); }
        { v8u t;
          #pragma unroll
          for (int g = 0; g < 8; ++g)
              t[g] = *(const unsigned int*)&Ps[w][ln16][32 + frag_k(g, kh)];
          ap1 = __builtin_bit_cast(v16bf, t); }
        #pragma unroll
        for (int dn = 0; dn < 4; ++dn) {
            v16bf bv0, bv1;
            { v8u t;
              #pragma unroll
              for (int g = 0; g < 8; ++g)
                  t[g] = *(const unsigned int*)&Vs[dn * 16 + ln16][frag_k(g, kh)];
              bv0 = __builtin_bit_cast(v16bf, t); }
            { v8u t;
              #pragma unroll
              for (int g = 0; g < 8; ++g)
                  t[g] = *(const unsigned int*)&Vs[dn * 16 + ln16][32 + frag_k(g, kh)];
              bv1 = __builtin_bit_cast(v16bf, t); }
            oc[dn] = __builtin_amdgcn_wmma_f32_16x16x32_bf16(false, ap0, false, bv0, (short)0, oc[dn], false, false);
            oc[dn] = __builtin_amdgcn_wmma_f32_16x16x32_bf16(false, ap1, false, bv1, (short)0, oc[dn], false, false);
        }
    }

    #pragma unroll
    for (int r = 0; r < 8; ++r) {
        float inv = (l_r[r] > 0.f) ? (1.f / l_r[r]) : 0.f;
        int qlocal = w * 16 + kh * 8 + r;
        int row = b * S_ + qb * 64 + qlocal;
        #pragma unroll
        for (int dn = 0; dn < 4; ++dn) {
            int d = dn * 16 + ln16;
            O[(size_t)row * HD_ + h * Dh_ + d] = f2bf(oc[dn][r] * inv);
        }
    }
}

// ---------------- host launcher ---------------------------------------------
extern "C" void kernel_launch(void* const* d_in, const int* in_sizes, int n_in,
                              void* d_out, int out_size, void* d_ws, size_t ws_size,
                              hipStream_t stream) {
    (void)in_sizes; (void)n_in; (void)out_size; (void)ws_size;
    const float* encoded  = (const float*)d_in[0];
    const float* srcmask  = (const float*)d_in[1];
    const int*   targets  = (const int*)  d_in[2];
    const float* table    = (const float*)d_in[3];
    const float* pos      = (const float*)d_in[4];
    const float* ln1_s    = (const float*)d_in[5];
    const float* ln1_b    = (const float*)d_in[6];
    const float* ln2_s    = (const float*)d_in[7];
    const float* ln2_b    = (const float*)d_in[8];
    const float* ln3_s    = (const float*)d_in[9];
    const float* ln3_b    = (const float*)d_in[10];
    const float* self_wq  = (const float*)d_in[11];
    const float* self_wk  = (const float*)d_in[12];
    const float* self_wv  = (const float*)d_in[13];
    const float* self_wo  = (const float*)d_in[14];
    const float* cross_wq = (const float*)d_in[15];
    const float* cross_wk = (const float*)d_in[16];
    const float* cross_wv = (const float*)d_in[17];
    const float* cross_wo = (const float*)d_in[18];
    const float* mlp_w1   = (const float*)d_in[19];
    const float* mlp_b1   = (const float*)d_in[20];
    const float* mlp_w2   = (const float*)d_in[21];
    const float* mlp_b2   = (const float*)d_in[22];
    const float* final_s  = (const float*)d_in[23];
    const float* final_b  = (const float*)d_in[24];
    const float* logit_w  = (const float*)d_in[25];
    const float* logit_b  = (const float*)d_in[26];
    float* out = (float*)d_out;

    char* ws = (char*)d_ws;
    size_t off = 0;
    auto take = [&](size_t bytes) -> void* {
        void* p = ws + off;
        off = (off + bytes + 255) & ~(size_t)255;
        return p;
    };
    float* y   = (float*)take((size_t)ROWS_ * E_  * 4);   // residual stream (f32)
    u16*   x   = (u16*)  take((size_t)ROWS_ * E_  * 2);   // LN output (bf16)
    u16*   q   = (u16*)  take((size_t)ROWS_ * HD_ * 2);
    u16*   k   = (u16*)  take((size_t)ROWS_ * HD_ * 2);
    u16*   v   = (u16*)  take((size_t)ROWS_ * HD_ * 2);
    u16*   o   = (u16*)  take((size_t)ROWS_ * HD_ * 2);
    u16*   mid = (u16*)  take((size_t)ROWS_ * M_  * 2);
    u16*   benc= (u16*)  take((size_t)ROWS_ * E_  * 2);
    // pre-transposed bf16 weights (B[N][K])
    u16* bwq_s = (u16*)take((size_t)L_*HD_*E_*2);
    u16* bwk_s = (u16*)take((size_t)L_*HD_*E_*2);
    u16* bwv_s = (u16*)take((size_t)L_*HD_*E_*2);
    u16* bwo_s = (u16*)take((size_t)L_*E_*HD_*2);
    u16* bwq_c = (u16*)take((size_t)L_*HD_*E_*2);
    u16* bwk_c = (u16*)take((size_t)L_*HD_*E_*2);
    u16* bwv_c = (u16*)take((size_t)L_*HD_*E_*2);
    u16* bwo_c = (u16*)take((size_t)L_*E_*HD_*2);
    u16* bw1   = (u16*)take((size_t)L_*M_*E_*2);
    u16* bw2   = (u16*)take((size_t)L_*E_*M_*2);
    u16* blog  = (u16*)take((size_t)V_*E_*2);
    unsigned char* masks = (unsigned char*)take(4 * NB_ * NB_);

    auto cvtT = [&](const float* in, u16* outp, int K, int N) {
        int n = K * N;
        cvt_t_kernel<<<(n + 255) / 256, 256, 0, stream>>>(in, outp, K, N);
    };

    build_masks_kernel<<<1, 1, 0, stream>>>(masks);
    // weight / encoder conversions (once per launch)
    for (int l = 0; l < L_; ++l) {
        cvtT(self_wq  + (size_t)l*E_*HD_, bwq_s + (size_t)l*HD_*E_, E_, HD_);
        cvtT(self_wk  + (size_t)l*E_*HD_, bwk_s + (size_t)l*HD_*E_, E_, HD_);
        cvtT(self_wv  + (size_t)l*E_*HD_, bwv_s + (size_t)l*HD_*E_, E_, HD_);
        cvtT(self_wo  + (size_t)l*HD_*E_, bwo_s + (size_t)l*E_*HD_, HD_, E_);
        cvtT(cross_wq + (size_t)l*E_*HD_, bwq_c + (size_t)l*HD_*E_, E_, HD_);
        cvtT(cross_wk + (size_t)l*E_*HD_, bwk_c + (size_t)l*HD_*E_, E_, HD_);
        cvtT(cross_wv + (size_t)l*E_*HD_, bwv_c + (size_t)l*HD_*E_, E_, HD_);
        cvtT(cross_wo + (size_t)l*HD_*E_, bwo_c + (size_t)l*E_*HD_, HD_, E_);
        cvtT(mlp_w1   + (size_t)l*E_*M_,  bw1   + (size_t)l*M_*E_,  E_, M_);
        cvtT(mlp_w2   + (size_t)l*M_*E_,  bw2   + (size_t)l*E_*M_,  M_, E_);
    }
    cvtT(logit_w, blog, E_, V_);
    {
        int n = ROWS_ * E_;
        cvt_kernel<<<(n + 255) / 256, 256, 0, stream>>>(encoded, benc, n);
    }
    embed_kernel<<<ROWS_, 128, 0, stream>>>(targets, table, pos, y);

    const float qs = 0.125f;      // 1/sqrt(D)
    dim3 ag(NB_, H_, B_);
    for (int l = 0; l < L_; ++l) {
        const unsigned char* smask = masks + (size_t)l * NB_ * NB_;
        const unsigned char* cmask = masks + (size_t)(2 + l) * NB_ * NB_;
        // ---- self attention ----
        layernorm_kernel<<<ROWS_, 128, 0, stream>>>(y, x, ln1_s + l*E_, ln1_b + l*E_);
        gemm_kernel<false,false,false,true><<<dim3(HD_/BN, ROWS_/BM), 256, 0, stream>>>(
            x, bwq_s + (size_t)l*HD_*E_, q, nullptr, nullptr, ROWS_, HD_, E_, qs);
        gemm_kernel<false,false,false,true><<<dim3(HD_/BN, ROWS_/BM), 256, 0, stream>>>(
            x, bwk_s + (size_t)l*HD_*E_, k, nullptr, nullptr, ROWS_, HD_, E_, 1.f);
        gemm_kernel<false,false,false,true><<<dim3(HD_/BN, ROWS_/BM), 256, 0, stream>>>(
            x, bwv_s + (size_t)l*HD_*E_, v, nullptr, nullptr, ROWS_, HD_, E_, 1.f);
        attn_kernel<<<ag, 128, 0, stream>>>(q, k, v, o, smask, targets, nullptr, 1);
        gemm_kernel<false,true,false,false><<<dim3(E_/BN, ROWS_/BM), 256, 0, stream>>>(
            o, bwo_s + (size_t)l*E_*HD_, y, nullptr, y, ROWS_, E_, HD_, 1.f);
        // ---- cross attention ----
        layernorm_kernel<<<ROWS_, 128, 0, stream>>>(y, x, ln2_s + l*E_, ln2_b + l*E_);
        gemm_kernel<false,false,false,true><<<dim3(HD_/BN, ROWS_/BM), 256, 0, stream>>>(
            x, bwq_c + (size_t)l*HD_*E_, q, nullptr, nullptr, ROWS_, HD_, E_, qs);
        gemm_kernel<false,false,false,true><<<dim3(HD_/BN, ROWS_/BM), 256, 0, stream>>>(
            benc, bwk_c + (size_t)l*HD_*E_, k, nullptr, nullptr, ROWS_, HD_, E_, 1.f);
        gemm_kernel<false,false,false,true><<<dim3(HD_/BN, ROWS_/BM), 256, 0, stream>>>(
            benc, bwv_c + (size_t)l*HD_*E_, v, nullptr, nullptr, ROWS_, HD_, E_, 1.f);
        attn_kernel<<<ag, 128, 0, stream>>>(q, k, v, o, cmask, targets, srcmask, 0);
        gemm_kernel<false,true,false,false><<<dim3(E_/BN, ROWS_/BM), 256, 0, stream>>>(
            o, bwo_c + (size_t)l*E_*HD_, y, nullptr, y, ROWS_, E_, HD_, 1.f);
        // ---- MLP ----
        layernorm_kernel<<<ROWS_, 128, 0, stream>>>(y, x, ln3_s + l*E_, ln3_b + l*E_);
        gemm_kernel<true,false,true,true><<<dim3(M_/BN, ROWS_/BM), 256, 0, stream>>>(
            x, bw1 + (size_t)l*M_*E_, mid, mlp_b1 + (size_t)l*M_, nullptr, ROWS_, M_, E_, 1.f);
        gemm_kernel<true,true,false,false><<<dim3(E_/BN, ROWS_/BM), 256, 0, stream>>>(
            mid, bw2 + (size_t)l*E_*M_, y, mlp_b2 + (size_t)l*E_, y, ROWS_, E_, M_, 1.f);
    }
    // ---- final LN + logits ----
    layernorm_kernel<<<ROWS_, 128, 0, stream>>>(y, x, final_s, final_b);
    gemm_kernel<true,false,false,false><<<dim3(V_/BN, ROWS_/BM), 256, 0, stream>>>(
        x, blog, out, logit_b, nullptr, ROWS_, V_, E_, 1.f);
}
